// MultiheadAttention_18966575579155
// MI455X (gfx1250) — matrix operand order repaired and checked by reading, not compiled, not run on hardware
//
#include <hip/hip_runtime.h>
#include <hip/hip_bf16.h>

// ---------------------------------------------------------------------------
// MultiheadAttention for MI455X (gfx1250): bf16 WMMA pipeline with
// async global->LDS double-buffered GEMMs (ASYNCcnt path).
//   L=256, B=32, E=1024, H=16, hd=64.  M = L*B = 8192 rows.
// ---------------------------------------------------------------------------

typedef __bf16 bf16_t;
typedef __attribute__((ext_vector_type(16))) __bf16 v16bf;
typedef __attribute__((ext_vector_type(8)))  __bf16 v8bf;
typedef __attribute__((ext_vector_type(4)))  __bf16 v4bf;
typedef __attribute__((ext_vector_type(8)))  float   v8f;
typedef __attribute__((ext_vector_type(4)))  float   v4f;

#define LN   256          // sequence length
#define BS   32           // batch
#define NH   16           // heads
#define HD   64           // head dim
#define ED   1024         // embed dim
#define MROWS (LN * BS)   // 8192
#define NHB   (BS * NH)   // 512 batch-heads

union AFragU { v16bf v; v8bf h[2]; };

// A-operand (16x32 bf16, row-major source): lane holds row (lane&15),
// two contiguous 8-elem chunks at K-offsets half*8 and 16+half*8.
__device__ inline v16bf load_a_frag(const bf16_t* base, int row_stride,
                                    int row, int kbase, int lane) {
    const int half = lane >> 4;
    const bf16_t* p = base + (size_t)row * row_stride + kbase;
    AFragU a;
    a.h[0] = *(const v8bf*)(p + half * 8);
    a.h[1] = *(const v8bf*)(p + 16 + half * 8);
    return a.v;
}

// B-operand (32x16 bf16) supplied as its TRANSPOSE (N x K row-major):
// lane holds col (lane&15), two 8-elem chunks at K-offsets half*8 and 16+half*8.
__device__ inline v16bf load_bt_frag(const bf16_t* base, int row_stride,
                                     int col, int kbase, int lane) {
    const int half = lane >> 4;
    const bf16_t* p = base + (size_t)col * row_stride + kbase;
    AFragU b;
    b.h[0] = *(const v8bf*)(p + half * 8);
    b.h[1] = *(const v8bf*)(p + 16 + half * 8);
    return b.v;
}

__device__ inline v8f wmma_bf16(v16bf a, v16bf b, v8f c) {
    return __builtin_amdgcn_wmma_f32_16x16x32_bf16(
        /*neg_a=*/false, a, /*neg_b=*/false, b,
        /*c_mod=*/(short)0, c, /*reuse_a=*/false, /*reuse_b=*/false);
}

// Async global -> LDS copy, 16B per lane (GLOBAL_LOAD_ASYNC_TO_LDS_B128,
// tracked by ASYNCcnt).  lds_off = LDS byte offset (low 32 bits of the
// generic shared-memory address, per ISA 10.2 aperture rules).
__device__ inline void async_g2l_b128(unsigned lds_off, const void* gaddr) {
    asm volatile("global_load_async_to_lds_b128 %0, %1, off"
                 :: "v"(lds_off), "v"(gaddr)
                 : "memory");
}

__device__ inline unsigned lds_off_of(const void* p) {
    return (unsigned)(unsigned long long)p;
}

// ---------------------------------------------------------------------------
// f32 -> bf16 cast, 4 elems/thread
// ---------------------------------------------------------------------------
__global__ __launch_bounds__(256)
void mha_cast_bf16(const float* __restrict__ src, bf16_t* __restrict__ dst, int count4) {
    int i = blockIdx.x * 256 + threadIdx.x;
    if (i < count4) {
        v4f x = ((const v4f*)src)[i];
        v4bf y;
        y.x = (bf16_t)x.x; y.y = (bf16_t)x.y;
        y.z = (bf16_t)x.z; y.w = (bf16_t)x.w;
        ((v4bf*)dst)[i] = y;
    }
}

// ---------------------------------------------------------------------------
// GEMM: C(M x 1024) = A(M x 1024, bf16) * Wt(1024 x 1024, bf16, N-major) + bias
//   Block: 256 thr = 8 waves; block tile 128(M) x 64(N); K-step 64.
//   A/B panels staged in LDS via async global->LDS, double buffered;
//   final K iteration peeled so the steady-state loop is branch-free.
// mode 0: q  -> scale & store (n,l,d) bf16
// mode 1: k  -> store (n,l,d) bf16
// mode 2: v  -> store transposed (n,d,l) bf16
// mode 3: out-projection -> store f32 row-major (== (L,B,E) flat)
// ---------------------------------------------------------------------------
__global__ __launch_bounds__(256)
void mha_gemm_kernel(const bf16_t* __restrict__ A, const bf16_t* __restrict__ Wt,
                     const float* __restrict__ bias, void* __restrict__ out,
                     int mode, float scale) {
    __shared__ bf16_t sA[2][128][64];   // 2 x 16 KB
    __shared__ bf16_t sB[2][64][64];    // 2 x  8 KB

    const int lane  = threadIdx.x & 31;
    const int wave  = threadIdx.x >> 5;
    const int half  = lane >> 4;
    const int rowg0 = blockIdx.x * 128;
    const int col0  = blockIdx.y * 64;

    const int rsub = lane >> 3;          // 0..3  (row within a 4-row group)
    const int eoff = (lane & 7) * 8;     // bf16 element offset (16B chunks)

    // Stage one 128x64 A panel + 64x64 B panel into LDS buffer `buf`.
    // Per wave: 4 async b128 for A (16 rows) + 2 for B (8 rows) = 6 issues.
    auto stage = [&](int kk, int buf) {
        #pragma unroll
        for (int i = 0; i < 4; ++i) {
            const int r = wave * 16 + i * 4 + rsub;
            const bf16_t* g = A + (size_t)(rowg0 + r) * ED + kk + eoff;
            async_g2l_b128(lds_off_of(&sA[buf][r][eoff]), g);
        }
        #pragma unroll
        for (int j = 0; j < 2; ++j) {
            const int r = wave * 8 + j * 4 + rsub;
            const bf16_t* g = Wt + (size_t)(col0 + r) * ED + kk + eoff;
            async_g2l_b128(lds_off_of(&sB[buf][r][eoff]), g);
        }
    };

    v8f acc[4];
    #pragma unroll
    for (int nt = 0; nt < 4; ++nt)
        acc[nt] = v8f{0.f, 0.f, 0.f, 0.f, 0.f, 0.f, 0.f, 0.f};

    auto compute = [&](int buf) {
        #pragma unroll
        for (int kc = 0; kc < 2; ++kc) {
            v16bf a = load_a_frag(&sA[buf][0][0], 64, wave * 16 + (lane & 15), kc * 32, lane);
            #pragma unroll
            for (int nt = 0; nt < 4; ++nt) {
                v16bf b = load_bt_frag(&sB[buf][0][0], 64, nt * 16 + (lane & 15), kc * 32, lane);
                acc[nt] = wmma_bf16(a, b, acc[nt]);
            }
        }
    };

    stage(0, 0);
    int buf = 0;
    #pragma unroll 1
    for (int kk = 0; kk < ED - 64; kk += 64) {
        stage(kk + 64, buf ^ 1);
        // 6 outstanding from the new issue; async loads complete in order, so
        // <=6 outstanding means this iteration's panel is resident.
        asm volatile("s_wait_asynccnt 0x6" ::: "memory");
        __syncthreads();
        compute(buf);
        __syncthreads();
        buf ^= 1;
    }
    // peeled tail: no further staging, drain ASYNCcnt fully
    asm volatile("s_wait_asynccnt 0x0" ::: "memory");
    __syncthreads();
    compute(buf);

    const int row0 = rowg0 + wave * 16;
    #pragma unroll
    for (int nt = 0; nt < 4; ++nt) {
        const int col  = col0 + nt * 16 + (lane & 15);
        const float bc = bias[col];
        #pragma unroll
        for (int e = 0; e < 8; ++e) {
            const int m = row0 + e + half * 8;
            float val = acc[nt][e] + bc;
            if (mode == 0) val *= scale;
            if (mode == 3) {
                ((float*)out)[(size_t)m * ED + col] = val;
            } else {
                const int l  = m >> 5;       // m / BS
                const int bb = m & 31;       // m % BS
                const int h  = col >> 6;     // col / HD
                const int d  = col & 63;     // col % HD
                const int n  = bb * NH + h;  // batch-major fused index
                bf16_t bv = (bf16_t)val;
                bf16_t* o = (bf16_t*)out;
                if (mode == 2) o[((size_t)n * HD + d) * LN + l] = bv;  // vt: (n,d,l)
                else           o[((size_t)n * LN + l) * HD + d] = bv;  // q,k: (n,l,d)
            }
        }
    }
}

// ---------------------------------------------------------------------------
// Attention core: per batch-head n, per wave a 16-row query strip.
//   S = q k^T ; S += rel_bias[n/32][|l-m|+256] ; S = sigmoid(S)*mask ;
//   ctx = S v  (staged through LDS to redistribute C-layout -> A-layout)
// Block: 128 threads = 4 waves -> 64 query rows; grid (512, 4).
// ---------------------------------------------------------------------------
__global__ __launch_bounds__(128)
void mha_attn_kernel(const bf16_t* __restrict__ Q, const bf16_t* __restrict__ K,
                     const bf16_t* __restrict__ VT, const float* __restrict__ mask,
                     const float* __restrict__ rel_bias, bf16_t* __restrict__ ctx) {
    __shared__ bf16_t s_scores[4][16][LN];   // 4 waves * 16 rows * 256 cols = 32 KB

    const int lane = threadIdx.x & 31;
    const int wave = threadIdx.x >> 5;
    const int half = lane >> 4;
    const int n    = blockIdx.x;
    const int row0 = blockIdx.y * 64 + wave * 16;

    const bf16_t* qn  = Q  + (size_t)n * LN * HD;
    const bf16_t* kn  = K  + (size_t)n * LN * HD;
    const bf16_t* vtn = VT + (size_t)n * HD * LN;
    const float*  mk  = mask + (size_t)n * LN * LN;
    // head-major repeat: bias row = rel_bias[n / B], window starts at +256
    const float*  rb  = rel_bias + (size_t)(n >> 5) * 512 + 256;

    // q strip A-fragments for the two K-steps (K = 64 = 2 x 32)
    v16bf aq0 = load_a_frag(qn, HD, row0 + (lane & 15), 0,  lane);
    v16bf aq1 = load_a_frag(qn, HD, row0 + (lane & 15), 32, lane);

    bf16_t (*srow)[LN] = s_scores[wave];

    for (int t = 0; t < 16; ++t) {                 // 16 key tiles of 16
        v8f acc = v8f{0.f, 0.f, 0.f, 0.f, 0.f, 0.f, 0.f, 0.f};
        v16bf b0 = load_bt_frag(kn, HD, t * 16 + (lane & 15), 0,  lane);
        acc = wmma_bf16(aq0, b0, acc);
        v16bf b1 = load_bt_frag(kn, HD, t * 16 + (lane & 15), 32, lane);
        acc = wmma_bf16(aq1, b1, acc);

        const int col = t * 16 + (lane & 15);
        #pragma unroll
        for (int e = 0; e < 8; ++e) {
            const int r  = e + half * 8;
            const int lq = row0 + r;
            const int dd = lq > col ? lq - col : col - lq;
            float v = acc[e] + rb[dd];
            v = 1.0f / (1.0f + __expf(-v));        // sigmoid gate
            v *= mk[(size_t)lq * LN + col];
            srow[r][col] = (bf16_t)v;
        }
    }
    __syncthreads();

    // ctx strip (16 x 64) = S (16 x 256) * v (256 x 64); vt is B^T layout.
    v8f accv[4];
    #pragma unroll
    for (int nt = 0; nt < 4; ++nt)
        accv[nt] = v8f{0.f, 0.f, 0.f, 0.f, 0.f, 0.f, 0.f, 0.f};

    for (int kk = 0; kk < LN; kk += 32) {
        v16bf a = load_a_frag(&srow[0][0], LN, lane & 15, kk, lane);
        #pragma unroll
        for (int nt = 0; nt < 4; ++nt) {
            v16bf b = load_bt_frag(vtn, LN, nt * 16 + (lane & 15), kk, lane);
            accv[nt] = wmma_bf16(a, b, accv[nt]);
        }
    }

    const int bb = n >> 4;     // n / NH
    const int h  = n & 15;     // n % NH
    #pragma unroll
    for (int nt = 0; nt < 4; ++nt) {
        #pragma unroll
        for (int e = 0; e < 8; ++e) {
            const int lq = row0 + e + half * 8;
            const int d  = nt * 16 + (lane & 15);
            ctx[((size_t)lq * BS + bb) * ED + h * HD + d] = (bf16_t)accv[nt][e];
        }
    }
}

// ---------------------------------------------------------------------------
// Host-side launch
// ---------------------------------------------------------------------------
extern "C" void kernel_launch(void* const* d_in, const int* in_sizes, int n_in,
                              void* d_out, int out_size, void* d_ws, size_t ws_size,
                              hipStream_t stream) {
    (void)in_sizes; (void)n_in; (void)out_size; (void)ws_size;

    const float* query  = (const float*)d_in[0];
    const float* amask  = (const float*)d_in[1];
    const float* q_w    = (const float*)d_in[2];
    const float* q_b    = (const float*)d_in[3];
    const float* k_w    = (const float*)d_in[4];
    const float* k_b    = (const float*)d_in[5];
    const float* v_w    = (const float*)d_in[6];
    const float* v_b    = (const float*)d_in[7];
    const float* out_w  = (const float*)d_in[8];
    const float* out_b  = (const float*)d_in[9];
    const float* rbias  = (const float*)d_in[10];

    char* ws = (char*)d_ws;
    const size_t SZ_X  = (size_t)MROWS * ED * sizeof(bf16_t);    // 16 MB
    const size_t SZ_W  = (size_t)ED * ED * sizeof(bf16_t);       //  2 MB
    const size_t SZ_T  = (size_t)NHB * LN * HD * sizeof(bf16_t); // 16 MB

    bf16_t* Xb  = (bf16_t*)(ws);
    bf16_t* WQ  = (bf16_t*)(ws + SZ_X);
    bf16_t* WK  = (bf16_t*)(ws + SZ_X + SZ_W);
    bf16_t* WV  = (bf16_t*)(ws + SZ_X + 2 * SZ_W);
    bf16_t* WO  = (bf16_t*)(ws + SZ_X + 3 * SZ_W);
    bf16_t* Qb  = (bf16_t*)(ws + SZ_X + 4 * SZ_W);
    bf16_t* Kb  = (bf16_t*)(ws + SZ_X + 4 * SZ_W + SZ_T);
    bf16_t* VTb = (bf16_t*)(ws + SZ_X + 4 * SZ_W + 2 * SZ_T);
    bf16_t* CTX = (bf16_t*)(ws + SZ_X + 4 * SZ_W + 3 * SZ_T);

    // 1) casts (4 elems/thread)
    mha_cast_bf16<<<(MROWS * ED / 4) / 256, 256, 0, stream>>>(query, Xb, MROWS * ED / 4);
    mha_cast_bf16<<<(ED * ED / 4) / 256, 256, 0, stream>>>(q_w,   WQ, ED * ED / 4);
    mha_cast_bf16<<<(ED * ED / 4) / 256, 256, 0, stream>>>(k_w,   WK, ED * ED / 4);
    mha_cast_bf16<<<(ED * ED / 4) / 256, 256, 0, stream>>>(v_w,   WV, ED * ED / 4);
    mha_cast_bf16<<<(ED * ED / 4) / 256, 256, 0, stream>>>(out_w, WO, ED * ED / 4);

    // 2) q/k/v projections (fused scale + layout epilogues)
    dim3 gGemm(MROWS / 128, ED / 64);   // (64, 16)
    const float scaling = 0.125f;       // 64^-0.5
    mha_gemm_kernel<<<gGemm, 256, 0, stream>>>(Xb, WQ, q_b, (void*)Qb,  0, scaling);
    mha_gemm_kernel<<<gGemm, 256, 0, stream>>>(Xb, WK, k_b, (void*)Kb,  1, 1.0f);
    mha_gemm_kernel<<<gGemm, 256, 0, stream>>>(Xb, WV, v_b, (void*)VTb, 2, 1.0f);

    // 3) attention core
    mha_attn_kernel<<<dim3(NHB, 4), 128, 0, stream>>>(Qb, Kb, VTb, amask, rbias, CTX);

    // 4) output projection -> f32 (L,B,E)
    mha_gemm_kernel<<<gGemm, 256, 0, stream>>>(CTX, WO, out_b, d_out, 3, 1.0f);
}
